// G_ATT_85332410237226
// MI455X (gfx1250) — matrix operand order repaired and checked by reading, not compiled, not run on hardware
//
#include <hip/hip_runtime.h>
#include <hip/hip_bf16.h>

typedef float    v8f  __attribute__((ext_vector_type(8)));
typedef _Float16 v4h  __attribute__((ext_vector_type(4)));
typedef _Float16 v8h  __attribute__((ext_vector_type(8)));
typedef _Float16 v16h __attribute__((ext_vector_type(16)));

#define B_   256
#define L_   512
#define E2_  512
#define D_   256

// ---------------------------------------------------------------------------
// Kernel T: Wg1T[e,k] = (f16) Wg1[k,e]  -- k-contiguous B operand for WMMA.
// 32x32 tiles through LDS, coalesced both directions.
// ---------------------------------------------------------------------------
__global__ __launch_bounds__(256) void transpose_kernel(const float* __restrict__ Wg1,
                                                        _Float16* __restrict__ Wg1T) {
    __shared__ float t[32][33];
    const int tx = threadIdx.x & 31;
    const int ty = threadIdx.x >> 5;          // 0..7
    const int k0 = blockIdx.x * 32;
    const int e0 = blockIdx.y * 32;
    #pragma unroll
    for (int r = ty; r < 32; r += 8)
        t[r][tx] = Wg1[(size_t)(k0 + r) * E2_ + e0 + tx];
    __syncthreads();
    #pragma unroll
    for (int r = ty; r < 32; r += 8)
        Wg1T[(size_t)(e0 + r) * E2_ + k0 + tx] = (_Float16)t[tx][r];
}

// ---------------------------------------------------------------------------
// Kernel A: qw[b,e] = sum_k C_target[b,k] * Wg2[k,e] + bias[e]
// ---------------------------------------------------------------------------
__global__ __launch_bounds__(512) void qw_kernel(const float* __restrict__ Ct,
                                                 const float* __restrict__ Wg2,
                                                 const float* __restrict__ bias,
                                                 float* __restrict__ qw) {
    __shared__ float ct[8 * E2_];
    const int b0 = blockIdx.x * 8;
    const int e  = threadIdx.x;
    #pragma unroll
    for (int r = 0; r < 8; ++r)
        ct[r * E2_ + e] = Ct[(size_t)(b0 + r) * E2_ + e];
    __syncthreads();

    float acc[8] = {};
    for (int k = 0; k < E2_; ++k) {
        const float w = Wg2[(size_t)k * E2_ + e];
        #pragma unroll
        for (int r = 0; r < 8; ++r) acc[r] += ct[r * E2_ + k] * w;
    }
    const float bv = bias[e];
    #pragma unroll
    for (int r = 0; r < 8; ++r)
        qw[(size_t)(b0 + r) * E2_ + e] = acc[r] + bv;
}

// ---------------------------------------------------------------------------
// Kernel B (WMMA f16->f32): score[b,l] = sum_e relu((C@Wg1)[b,l,e]+qw[b,e])*Wg0[e]
// Block: b fixed, 64 L-rows as f16 in LDS.  8 waves = 4 E-groups x 2 parities.
// Each wave: M-unroll 4 x N-unroll 2 -> one A-fragment set feeds 8 independent
// v_wmma_f32_16x16x32_f16; Wg1T read from L2 once per 64 rows (1 GB total),
// A-LDS traffic ~2 GB -- balanced vs matrix pipe (~14us) and HBM floor (~29us).
// ---------------------------------------------------------------------------
#define LTILE 64
#define HSTRIDE 520   // halves; 260 dwords = 4 mod 64 -> conflict-free b128 reads

__global__ __launch_bounds__(256) void score_kernel(const float* __restrict__ C,
                                                    const _Float16* __restrict__ Wg1T,
                                                    const float* __restrict__ qw,
                                                    const float* __restrict__ Wg0,
                                                    const int*   __restrict__ mask,
                                                    float* __restrict__ score) {
    __shared__ _Float16 ldsC[LTILE * HSTRIDE];   // ~65 KB
    __shared__ float    ldsS[LTILE];

    const int blk = blockIdx.x;           // 2048 blocks
    const int b   = blk >> 3;
    const int l0  = (blk & 7) * LTILE;
    const int tid = threadIdx.x;

    // Stage C[b, l0:l0+64, :] into LDS as f16, coalesced f32x4 -> f16x4.
    const float* Cb = C + ((size_t)b * L_ + l0) * E2_;
    for (int i = tid * 4; i < LTILE * E2_; i += 256 * 4) {
        const int row = i >> 9;
        const int col = i & 511;
        const float4 v = *(const float4*)(Cb + (size_t)row * E2_ + col);
        v4h h;
        h.x = (_Float16)v.x; h.y = (_Float16)v.y;
        h.z = (_Float16)v.z; h.w = (_Float16)v.w;
        *(v4h*)&ldsC[row * HSTRIDE + col] = h;
    }
    if (tid < LTILE) ldsS[tid] = 0.0f;
    __syncthreads();

    const int wave = tid >> 5;     // 0..7
    const int lane = tid & 31;
    const int wq   = wave & 3;     // 32-col E group within each 128-col chunk
    const int wh   = wave >> 2;    // estep parity
    const int n    = lane & 15;    // N col / A-row index
    const int g    = lane >> 4;    // K-group selector (A/B), M-half (C/D)

    float sacc[4][8] = {};

    for (int es = wh; es < 4; es += 2) {         // 2 esteps/wave, 128 cols each
        const int ea = es * 128 + wq * 32 + n;   // N-half 0
        const int eb = ea + 16;                  // N-half 1
        v8f c0[4] = {}, c1[4] = {};
        for (int k0 = 0; k0 < E2_; k0 += 32) {
            // B fragments: contiguous 16 halves (K=16g..16g+15) per lane.
            const v16h b0 = *(const v16h*)&Wg1T[(size_t)ea * E2_ + k0 + 16 * g];
            const v16h b1 = *(const v16h*)&Wg1T[(size_t)eb * E2_ + k0 + 16 * g];
            #pragma unroll
            for (int mt = 0; mt < 4; ++mt) {
                const int row = mt * 16 + n;
                const v8h lo = *(const v8h*)&ldsC[row * HSTRIDE + k0 + 16 * g];
                const v8h hi = *(const v8h*)&ldsC[row * HSTRIDE + k0 + 16 * g + 8];
                const v16h a = __builtin_shufflevector(lo, hi,
                        0, 1, 2, 3, 4, 5, 6, 7, 8, 9, 10, 11, 12, 13, 14, 15);
                c0[mt] = __builtin_amdgcn_wmma_f32_16x16x32_f16(
                             false, a, false, b0, (short)0, c0[mt], false, false);
                c1[mt] = __builtin_amdgcn_wmma_f32_16x16x32_f16(
                             false, a, false, b1, (short)0, c1[mt], false, false);
            }
        }
        // Epilogue: +qw, relu, dot with Wg0 for both N-halves.
        const float qv0 = qw[(size_t)b * E2_ + ea];
        const float w00 = Wg0[ea];
        const float qv1 = qw[(size_t)b * E2_ + eb];
        const float w01 = Wg0[eb];
        #pragma unroll
        for (int mt = 0; mt < 4; ++mt) {
            #pragma unroll
            for (int r = 0; r < 8; ++r) {
                float h0 = c0[mt][r] + qv0; h0 = h0 > 0.0f ? h0 : 0.0f;
                float h1 = c1[mt][r] + qv1; h1 = h1 > 0.0f ? h1 : 0.0f;
                sacc[mt][r] += h0 * w00 + h1 * w01;
            }
        }
    }

    // Reduce each 16-lane half (its 16 E columns), accumulate per-row in LDS.
    #pragma unroll
    for (int mt = 0; mt < 4; ++mt) {
        #pragma unroll
        for (int r = 0; r < 8; ++r) {
            float v = sacc[mt][r];
            v += __shfl_xor(v, 8, 16);
            v += __shfl_xor(v, 4, 16);
            v += __shfl_xor(v, 2, 16);
            v += __shfl_xor(v, 1, 16);
            if ((lane & 15) == 0)
                atomicAdd(&ldsS[mt * 16 + r + 8 * g], v);
        }
    }
    __syncthreads();

    if (tid < LTILE) {
        const int l = l0 + tid;
        const int m = mask[(size_t)b * L_ + l];
        score[(size_t)b * L_ + l] = (m != 0) ? ldsS[tid] : -1e30f;
    }
}

// ---------------------------------------------------------------------------
// Kernel C: alpha-entmax bisection over L per batch row (matches reference).
// ---------------------------------------------------------------------------
__device__ __forceinline__ float block_sum(float v, float* buf) {
    #pragma unroll
    for (int m = 16; m >= 1; m >>= 1) v += __shfl_xor(v, m, 32);
    const int wid = threadIdx.x >> 5;
    __syncthreads();
    if ((threadIdx.x & 31) == 0) buf[wid] = v;
    __syncthreads();
    if (threadIdx.x == 0) {
        float s = 0.0f;
        #pragma unroll
        for (int i = 0; i < 16; ++i) s += buf[i];
        buf[16] = s;
    }
    __syncthreads();
    return buf[16];
}

__device__ __forceinline__ float block_max(float v, float* buf) {
    #pragma unroll
    for (int m = 16; m >= 1; m >>= 1) v = fmaxf(v, __shfl_xor(v, m, 32));
    const int wid = threadIdx.x >> 5;
    __syncthreads();
    if ((threadIdx.x & 31) == 0) buf[wid] = v;
    __syncthreads();
    if (threadIdx.x == 0) {
        float s = buf[0];
        #pragma unroll
        for (int i = 1; i < 16; ++i) s = fmaxf(s, buf[i]);
        buf[16] = s;
    }
    __syncthreads();
    return buf[16];
}

__device__ __forceinline__ float pow_pos(float t, float inv) {
    return t > 0.0f ? exp2f(inv * log2f(t)) : 0.0f;
}

__global__ __launch_bounds__(512) void entmax_kernel(const float* __restrict__ score,
                                                     const float* __restrict__ alpha,
                                                     float* __restrict__ p) {
    __shared__ float red[17];
    const int b = blockIdx.x;
    const int l = threadIdx.x;

    const float am1 = alpha[0] - 1.0f;
    const float inv = 1.0f / am1;
    const float xs  = score[(size_t)b * L_ + l] * am1;

    const float mx     = block_max(xs, red);
    float       tau_lo = mx - 1.0f;
    const float tau_hi = mx - exp2f(-9.0f * am1);   // (1/512)^am1
    const float f_lo   = block_sum(pow_pos(xs - tau_lo, inv), red) - 1.0f;

    float dm = tau_hi - tau_lo;
    for (int it = 0; it < 50; ++it) {
        dm *= 0.5f;
        const float tau_m = tau_lo + dm;
        const float f_m   = block_sum(pow_pos(xs - tau_m, inv), red) - 1.0f;
        if (f_m * f_lo >= 0.0f) tau_lo = tau_m;     // uniform across block
    }
    const float pv   = pow_pos(xs - (tau_lo + 0.5f * dm), inv);
    const float psum = block_sum(pv, red);
    p[(size_t)b * L_ + l] = pv / psum;
}

// ---------------------------------------------------------------------------
// Kernel D: attn[b,e] = sum_l p[b,l] * C[b,l,e]   (bandwidth bound, coalesced)
// ---------------------------------------------------------------------------
__global__ __launch_bounds__(512) void attn_kernel(const float* __restrict__ C,
                                                   const float* __restrict__ p,
                                                   float* __restrict__ attn) {
    __shared__ float lp[L_];
    const int b = blockIdx.x;
    const int e = threadIdx.x;
    lp[e] = p[(size_t)b * L_ + e];
    __syncthreads();
    const float* Cb = C + (size_t)b * L_ * E2_;
    float acc = 0.0f;
    #pragma unroll 4
    for (int l = 0; l < L_; ++l)
        acc += lp[l] * Cb[(size_t)l * E2_ + e];
    attn[(size_t)b * E2_ + e] = acc;
}

// ---------------------------------------------------------------------------
// Kernel E: out[b,d] = sum_e attn[b,e] * X[b,e,d]  (bandwidth bound, coalesced)
// ---------------------------------------------------------------------------
__global__ __launch_bounds__(256) void out_kernel(const float* __restrict__ X,
                                                  const float* __restrict__ attn,
                                                  float* __restrict__ out) {
    __shared__ float la[E2_];
    const int b = blockIdx.x;
    const int d = threadIdx.x;
    la[d]       = attn[(size_t)b * E2_ + d];
    la[256 + d] = attn[(size_t)b * E2_ + 256 + d];
    __syncthreads();
    const float* Xb = X + (size_t)b * E2_ * D_;
    float acc = 0.0f;
    #pragma unroll 4
    for (int e = 0; e < E2_; ++e)
        acc += la[e] * Xb[(size_t)e * D_ + d];
    out[(size_t)b * D_ + d] = acc;
}

// ---------------------------------------------------------------------------
extern "C" void kernel_launch(void* const* d_in, const int* in_sizes, int n_in,
                              void* d_out, int out_size, void* d_ws, size_t ws_size,
                              hipStream_t stream) {
    const float* C_target = (const float*)d_in[0];   // [256,512]
    const float* C        = (const float*)d_in[1];   // [256,512,512]
    const float* X        = (const float*)d_in[2];   // [256,512,256]
    const float* alpha    = (const float*)d_in[3];   // [1]
    const int*   mask     = (const int*)  d_in[4];   // [256,512]
    const float* Wg1      = (const float*)d_in[5];   // [512,512]
    const float* Wg2      = (const float*)d_in[6];   // [512,512]
    const float* Wg0      = (const float*)d_in[7];   // [512,1]
    const float* bias     = (const float*)d_in[8];   // [512]
    float*       out      = (float*)d_out;           // [256,256]

    // Workspace: 4 x 512 KB f32 + 512 KB f16 Wg1T.
    float*    qw    = (float*)d_ws;            // [256,512]
    float*    score = qw    + B_ * E2_;        // [256,512]
    float*    p     = score + B_ * L_;         // [256,512]
    float*    attn  = p     + B_ * L_;         // [256,512]
    _Float16* Wg1T  = (_Float16*)(attn + B_ * E2_);   // [512,512] f16

    transpose_kernel<<<dim3(16, 16), 256, 0, stream>>>(Wg1, Wg1T);
    qw_kernel    <<<B_ / 8, 512, 0, stream>>>(C_target, Wg2, bias, qw);
    score_kernel <<<B_ * (L_ / LTILE), 256, 0, stream>>>(C, Wg1T, qw, Wg0, mask, score);
    entmax_kernel<<<B_, 512, 0, stream>>>(score, alpha, p);
    attn_kernel  <<<B_, 512, 0, stream>>>(C, p, attn);
    out_kernel   <<<B_, 256, 0, stream>>>(X, attn, out);
}
